// ES_MOE_11390253269274
// MI455X (gfx1250) — matrix operand order repaired and checked by reading, not compiled, not run on hardware
//
#include <hip/hip_runtime.h>
#include <hip/hip_bf16.h>

typedef _Float16 h16v __attribute__((ext_vector_type(16)));
typedef _Float16 h8v  __attribute__((ext_vector_type(8)));
typedef float    f8v  __attribute__((ext_vector_type(8)));

#define B_    32
#define C_    64
#define H_    160
#define W_    160
#define E_    3
#define RED_  8
#define TILE  16
#define HALO  3
#define XT    (TILE + 2*HALO)   /* 22 rows per channel */
#define XSTR  24                /* padded x-row stride (halfs): 48B -> 16B-aligned b128 */
#define NPIX  (TILE*TILE)       /* 256 */
#define YS_STR 72               /* padded row stride (halfs) */
#define PW_STR 72
#define BN_EPS 1e-5f

__device__ __forceinline__ float silu_f(float v) {
    return v / (1.f + __expf(-v));
}

// ---------------- routing: global average pool ----------------
__global__ void moe_pool_kernel(const float* __restrict__ x, float* __restrict__ pooled) {
    __shared__ float red[256];
    const int bc = blockIdx.x;            // b*C + c, 0..2047
    const int t  = threadIdx.x;
    const float* p = x + (size_t)bc * (H_ * W_);
    float s = 0.f;
    for (int i = t; i < H_ * W_; i += 256) s += p[i];
    red[t] = s;
    __syncthreads();
    for (int st = 128; st > 0; st >>= 1) {
        if (t < st) red[t] += red[t + st];
        __syncthreads();
    }
    if (t == 0) pooled[bc] = red[0] * (1.f / (float)(H_ * W_));
}

// ---------------- routing: MLP + top-2 softmax ----------------
__global__ void moe_route_kernel(const float* __restrict__ pooled,
                                 const float* __restrict__ w1, const float* __restrict__ b1,
                                 const float* __restrict__ w2, const float* __restrict__ b2,
                                 float* __restrict__ rw) {
    const int b = threadIdx.x;
    if (b >= B_) return;
    float h[RED_];
    #pragma unroll
    for (int r = 0; r < RED_; ++r) {
        float a = b1[r];
        for (int c = 0; c < C_; ++c) a += w1[r * C_ + c] * pooled[b * C_ + c];
        h[r] = silu_f(a);
    }
    float lg[E_];
    #pragma unroll
    for (int e = 0; e < E_; ++e) {
        float a = b2[e];
        #pragma unroll
        for (int r = 0; r < RED_; ++r) a += w2[e * RED_ + r] * h[r];
        lg[e] = a;
    }
    // top-2 (first-occurrence tie break, matches lax.top_k)
    int i0 = 0;
    if (lg[1] > lg[i0]) i0 = 1;
    if (lg[2] > lg[i0]) i0 = 2;
    int i1 = -1; float v1 = -3.4e38f;
    #pragma unroll
    for (int e = 0; e < E_; ++e)
        if (e != i0 && lg[e] > v1) { v1 = lg[e]; i1 = e; }
    float e1 = __expf(v1 - lg[i0]);
    float inv = 1.f / (1.f + e1);
    float t0 = inv, t1 = e1 * inv;
    #pragma unroll
    for (int e = 0; e < E_; ++e)
        rw[b * E_ + e] = (e == i0) ? t0 : ((e == i1) ? t1 : 0.f);
}

// ---------------- fused experts: depthwise + WMMA pointwise + BN/SiLU mix ----------------
__global__ __launch_bounds__(256)
void moe_main_kernel(const float* __restrict__ x,
                     const float* __restrict__ dw0,
                     const float* __restrict__ dw1,
                     const float* __restrict__ dw2,
                     const float* __restrict__ pw,
                     const float* __restrict__ bn_gamma,
                     const float* __restrict__ bn_beta,
                     const float* __restrict__ norm_gamma,
                     const float* __restrict__ norm_beta,
                     const float* __restrict__ rw,
                     float* __restrict__ out)
{
    __shared__ _Float16 xs[C_ * XT * XSTR];        // 67584 B  f16 input tile (+halo), padded rows
    __shared__ _Float16 ys[NPIX * YS_STR];         // 36864 B  depthwise out, [pixel][C]
    __shared__ _Float16 pws[C_ * PW_STR];          //  9216 B  expert pointwise weights f16
    __shared__ float    dws[C_ * 49];              // 12544 B  expert depthwise weights
    __shared__ float    sscale[E_ * C_], sbeta[E_ * C_];
    __shared__ float    nscale[C_], nbeta[C_];

    const int t   = threadIdx.x;
    const int b   = blockIdx.z;
    const int ty0 = blockIdx.y * TILE;
    const int tx0 = blockIdx.x * TILE;

    // stage BN affines (eval: mean=0, var=1)
    const float invs = rsqrtf(1.f + BN_EPS);
    for (int i = t; i < E_ * C_; i += 256) {
        sscale[i] = bn_gamma[i] * invs;
        sbeta[i]  = bn_beta[i];
    }
    if (t < C_) {
        nscale[t] = norm_gamma[t] * invs;
        nbeta[t]  = norm_beta[t];
    }

    // stage x tile with halo, zero-padded at image borders (pad columns 22..23 -> 0)
    for (int i = t; i < C_ * XT * XSTR; i += 256) {
        int c  = i / (XT * XSTR);
        int r  = i - c * (XT * XSTR);
        int iy = r / XSTR, ix = r - iy * XSTR;
        int gy = ty0 - HALO + iy;
        int gx = tx0 - HALO + ix;
        float v = 0.f;
        if (ix < XT && gy >= 0 && gy < H_ && gx >= 0 && gx < W_)
            v = x[(((size_t)b * C_ + c) * H_ + gy) * W_ + gx];
        xs[i] = (_Float16)v;
    }

    const int lane  = t & 31;
    const int wid   = t >> 5;          // 8 waves
    const int row16 = lane & 15;
    const int hi    = lane >> 4;       // half-wave select
    const int nt0   = wid * 2;         // this wave's two N-tiles (= tile rows py)

    f8v macc[4][2];
    #pragma unroll
    for (int mt = 0; mt < 4; ++mt)
        #pragma unroll
        for (int n = 0; n < 2; ++n)
            #pragma unroll
            for (int j = 0; j < 8; ++j) macc[mt][n][j] = 0.f;

    const float* dwp_[3] = { dw0, dw1, dw2 };

    #pragma unroll
    for (int e = 0; e < E_; ++e) {
        const int k  = (e == 0) ? 3 : (e == 1) ? 5 : 7;
        const int kk = k * k;
        const int pk = (k - 1) >> 1;
        __syncthreads();   // prior iteration's LDS reads complete

        // stage expert depthwise + pointwise weights
        const float* dwe = dwp_[e];
        for (int i = t; i < C_ * kk; i += 256) dws[i] = dwe[i];
        for (int i = t; i < C_ * C_; i += 256)
            pws[(i >> 6) * PW_STR + (i & 63)] = (_Float16)pw[e * C_ * C_ + i];
        __syncthreads();

        // depthwise conv: each task = (channel, tile row) strip of 16 output pixels.
        // Input row loaded once per tap-row via aligned b128s and reused for k*16 FMAs.
        for (int task = 0; task < 4; ++task) {
            const int idx = t + task * 256;   // 0..1023
            const int c   = idx >> 4;         // channel
            const int py  = idx & 15;         // output row in tile
            const float* wb = &dws[c * kk];
            float acc[TILE];
            #pragma unroll
            for (int px = 0; px < TILE; ++px) acc[px] = 0.f;

            for (int dy = 0; dy < k; ++dy) {
                const int iy = py + HALO - pk + dy;
                const h8v* xv = (const h8v*)&xs[(c * XT + iy) * XSTR];
                union { h8v v[3]; _Float16 h[24]; } ux;
                ux.v[0] = xv[0]; ux.v[1] = xv[1]; ux.v[2] = xv[2];
                float xrow[XT];
                #pragma unroll
                for (int i = 0; i < XT; ++i) xrow[i] = (float)ux.h[i];

                #pragma unroll
                for (int dx = 0; dx < 7; ++dx) {
                    if (dx < k) {
                        const float w = wb[dy * k + dx];
                        const int base = HALO - pk + dx;
                        #pragma unroll
                        for (int px = 0; px < TILE; ++px)
                            acc[px] = fmaf(xrow[base + px], w, acc[px]);
                    }
                }
            }
            #pragma unroll
            for (int px = 0; px < TILE; ++px)
                ys[((py << 4) + px) * YS_STR + c] = (_Float16)acc[px];
        }
        __syncthreads();

        // pointwise 1x1 as GEMM: out[o,p] = sum_c pw[o,c] * ys[p,c]
        f8v ea[4][2];
        #pragma unroll
        for (int mt = 0; mt < 4; ++mt)
            #pragma unroll
            for (int n = 0; n < 2; ++n)
                #pragma unroll
                for (int j = 0; j < 8; ++j) ea[mt][n][j] = 0.f;

        #pragma unroll
        for (int kt = 0; kt < 2; ++kt) {
            // B fragments: lane holds column p, contiguous K run (k = 16*hi + s, same order as A)
            h16v bf[2];
            #pragma unroll
            for (int n = 0; n < 2; ++n) {
                const int p = (nt0 + n) * 16 + row16;
                const _Float16* bp = &ys[p * YS_STR + kt * 32 + hi * 16];
                union { h8v h8[2]; h16v h; } ub;
                ub.h8[0] = *(const h8v*)(bp);
                ub.h8[1] = *(const h8v*)(bp + 8);
                bf[n] = ub.h;
            }
            #pragma unroll
            for (int mt = 0; mt < 4; ++mt) {
                // A fragment: lanes 0-15 -> K{0..15}, lanes 16-31 -> K{16..31} (matches B)
                const _Float16* ap = &pws[(mt * 16 + row16) * PW_STR + kt * 32 + hi * 16];
                union { h8v h8[2]; h16v h; } ua;
                ua.h8[0] = *(const h8v*)(ap);
                ua.h8[1] = *(const h8v*)(ap + 8);
                h16v af = ua.h;
                ea[mt][0] = __builtin_amdgcn_wmma_f32_16x16x32_f16(
                    false, af, false, bf[0], (short)0, ea[mt][0], false, false);
                ea[mt][1] = __builtin_amdgcn_wmma_f32_16x16x32_f16(
                    false, af, false, bf[1], (short)0, ea[mt][1], false, false);
            }
        }

        // per-expert BN (eval) + SiLU + routing-weighted accumulate
        const float we = rw[b * E_ + e];
        #pragma unroll
        for (int mt = 0; mt < 4; ++mt) {
            #pragma unroll
            for (int j = 0; j < 8; ++j) {
                const int o  = mt * 16 + j + hi * 8;
                const float sc = sscale[e * C_ + o];
                const float bt = sbeta[e * C_ + o];
                #pragma unroll
                for (int n = 0; n < 2; ++n) {
                    float y = ea[mt][n][j] * sc + bt;
                    macc[mt][n][j] += we * silu_f(y);
                }
            }
        }
    }

    // final BN + SiLU, coalesced store (lane -> px)
    #pragma unroll
    for (int mt = 0; mt < 4; ++mt) {
        #pragma unroll
        for (int n = 0; n < 2; ++n) {
            const int py = nt0 + n;
            const int gx = tx0 + row16;
            #pragma unroll
            for (int j = 0; j < 8; ++j) {
                const int o = mt * 16 + j + hi * 8;
                float y = macc[mt][n][j] * nscale[o] + nbeta[o];
                out[(((size_t)b * C_ + o) * H_ + (ty0 + py)) * W_ + gx] = silu_f(y);
            }
        }
    }
}

extern "C" void kernel_launch(void* const* d_in, const int* in_sizes, int n_in,
                              void* d_out, int out_size, void* d_ws, size_t ws_size,
                              hipStream_t stream) {
    (void)in_sizes; (void)n_in; (void)out_size; (void)ws_size;
    const float* x    = (const float*)d_in[0];
    const float* w1   = (const float*)d_in[1];
    const float* b1   = (const float*)d_in[2];
    const float* w2   = (const float*)d_in[3];
    const float* b2   = (const float*)d_in[4];
    const float* dw0  = (const float*)d_in[5];
    const float* dw1  = (const float*)d_in[6];
    const float* dw2  = (const float*)d_in[7];
    const float* pw   = (const float*)d_in[8];
    const float* bn_g = (const float*)d_in[9];
    const float* bn_b = (const float*)d_in[10];
    const float* ng   = (const float*)d_in[11];
    const float* nb   = (const float*)d_in[12];
    float* out = (float*)d_out;

    float* pooled = (float*)d_ws;            // B*C floats
    float* rw     = pooled + B_ * C_;        // B*E floats

    moe_pool_kernel<<<B_ * C_, 256, 0, stream>>>(x, pooled);
    moe_route_kernel<<<1, 32, 0, stream>>>(pooled, w1, b1, w2, b2, rw);

    dim3 grid(W_ / TILE, H_ / TILE, B_);
    moe_main_kernel<<<grid, 256, 0, stream>>>(x, dw0, dw1, dw2, pw,
                                              bn_g, bn_b, ng, nb, rw, out);
}
